// DocumentGraph_69475390980375
// MI455X (gfx1250) — compile-verified, matrix operand order checked by reading, not run on hardware
//
#include <hip/hip_runtime.h>
#include <hip/hip_fp16.h>

typedef __attribute__((ext_vector_type(16))) _Float16 v16h;
typedef __attribute__((ext_vector_type(8)))  float    v8f;

#define B_ 64
#define E_ 512
#define N_ 1024
#define F_ 128
#define ALPHA_ 0.2f
#define LDSROW 40   // 32 data halves + 8 pad halves: 80B row, bank-conflict-free

union V16 { uint4 u[2]; v16h h; };

__device__ __forceinline__ float lrelu(float v) { return v > 0.f ? v : ALPHA_ * v; }

__device__ __forceinline__ float wred_sum(float v) {
  v += __shfl_xor(v, 1, 32);  v += __shfl_xor(v, 2, 32);
  v += __shfl_xor(v, 4, 32);  v += __shfl_xor(v, 8, 32);
  v += __shfl_xor(v, 16, 32);
  return v;
}

// per-halfword keep-mask from two bits of a 32-bit mask word
__device__ __forceinline__ unsigned mask2(unsigned bits, int b0) {
  return (((bits >> b0) & 1u) ? 0x0000FFFFu : 0u) |
         (((bits >> (b0 + 1)) & 1u) ? 0xFFFF0000u : 0u);
}

// ---------------------------------------------------------------------------
// K0: fold W2/W3 into attention vectors.  vbuf = [v1(128) | v2(128) | v3(128) | c1]
// ---------------------------------------------------------------------------
__global__ void k_prep(const float* __restrict__ W2, const float* __restrict__ W3,
                       const float* __restrict__ wc, const float* __restrict__ a1,
                       const float* __restrict__ a2, float* __restrict__ vbuf) {
  int t = threadIdx.x;                       // 128 threads
  float s1v = 0.f, s2v = 0.f, s3v = 0.f;
  for (int g = 0; g < F_; ++g) {
    s1v += W2[t * F_ + g] * a1[F_ + g];
    s2v += W2[t * F_ + g] * a2[g];
    s3v += W3[t * F_ + g] * a2[F_ + g];
  }
  vbuf[t] = s1v; vbuf[F_ + t] = s2v; vbuf[2 * F_ + t] = s3v;
  if (t == 0) {
    float c = 0.f;
    for (int g = 0; g < F_; ++g) c += wc[g] * a1[g];
    vbuf[3 * F_] = c;
  }
}

// ---------------------------------------------------------------------------
// K1: embedding gather -> xT (f16, f-major [B][F][N]); node scores s1, sn.
// ---------------------------------------------------------------------------
__global__ void k_gather(const int* __restrict__ inputs, const float* __restrict__ emb,
                         const float* __restrict__ vbuf, _Float16* __restrict__ xT,
                         float* __restrict__ s1, float* __restrict__ sn) {
  int wid  = (blockIdx.x * blockDim.x + threadIdx.x) >> 5;
  int lane = threadIdx.x & 31;
  if (wid >= B_ * N_) return;
  int b = wid / N_, n = wid % N_;
  int idx = inputs[wid];
  float4 xr = *(const float4*)(emb + (size_t)idx * F_ + lane * 4);
  float4 w1 = *(const float4*)(vbuf + lane * 4);
  float4 w2 = *(const float4*)(vbuf + F_ + lane * 4);
  float d1 = xr.x * w1.x + xr.y * w1.y + xr.z * w1.z + xr.w * w1.w;
  float d2 = xr.x * w2.x + xr.y * w2.y + xr.z * w2.z + xr.w * w2.w;
  d1 = wred_sum(d1);
  d2 = wred_sum(d2);
  _Float16* base = xT + (size_t)b * F_ * N_ + (size_t)(lane * 4) * N_ + n;
  base[0]        = (_Float16)xr.x;
  base[N_]       = (_Float16)xr.y;
  base[2 * N_]   = (_Float16)xr.z;
  base[3 * N_]   = (_Float16)xr.w;
  if (lane == 0) {
    float c1 = vbuf[3 * F_];
    s1[wid] = lrelu(c1 + d1);
    sn[wid] = d2;
  }
}

// ---------------------------------------------------------------------------
// K2: per-batch stable exp: p = exp(s1 - max_b) stored f16
// ---------------------------------------------------------------------------
__global__ void k_pmax(const float* __restrict__ s1, _Float16* __restrict__ ph) {
  __shared__ float red[256];
  int b = blockIdx.x;
  const float* s = s1 + b * N_;
  float m = -1e30f;
  for (int n = threadIdx.x; n < N_; n += 256) m = fmaxf(m, s[n]);
  red[threadIdx.x] = m; __syncthreads();
  for (int o = 128; o > 0; o >>= 1) {
    if (threadIdx.x < (unsigned)o) red[threadIdx.x] = fmaxf(red[threadIdx.x], red[threadIdx.x + o]);
    __syncthreads();
  }
  m = red[0];
  for (int n = threadIdx.x; n < N_; n += 256) ph[b * N_ + n] = (_Float16)__expf(s[n] - m);
}

// ---------------------------------------------------------------------------
// K3: single pass over HT (134 MB) -> two bitmasks:
//   mask  [B][E][32 words]  (bit n%32)   via wave32 ballot (coalesced reads)
//   maskT [B][N][16 words]  (bit e%32)   via LDS aggregation (32 waves = 32 edges)
// ---------------------------------------------------------------------------
__global__ void k_mask(const int* __restrict__ HT, unsigned* __restrict__ mask,
                       unsigned* __restrict__ maskT) {
  __shared__ unsigned ldsT[N_];
  int b = blockIdx.x, eb = blockIdx.y;                // eb in [0, E/32)
  int w = threadIdx.x >> 5, lane = threadIdx.x & 31;  // 32 waves / WG
  ldsT[threadIdx.x] = 0u;
  __syncthreads();
  int e = eb * 32 + w;
  const int* row = HT + ((size_t)b * E_ + e) * N_;
  for (int it = 0; it < 32; ++it) {
    int n = it * 32 + lane;
    bool v = row[n] > 0;
    unsigned bal = (unsigned)__ballot(v);
    if (lane == 0) mask[((size_t)b * E_ + e) * 32 + it] = bal;
    if (v) atomicOr(&ldsT[n], 1u << w);
  }
  __syncthreads();
  for (int n = threadIdx.x; n < N_; n += blockDim.x)
    maskT[((size_t)b * N_ + n) * (E_ / 32) + eb] = ldsT[n];
}

// ---------------------------------------------------------------------------
// K4: softmax denominators per (b,e):  denom = sum_{n in edge} p[n]
// ---------------------------------------------------------------------------
__global__ void k_denom(const unsigned* __restrict__ mask, const _Float16* __restrict__ ph,
                        float* __restrict__ denom) {
  int wid  = (blockIdx.x * blockDim.x + threadIdx.x) >> 5;
  int lane = threadIdx.x & 31;
  if (wid >= B_ * E_) return;
  int b = wid / E_;
  unsigned bits = mask[(size_t)wid * 32 + lane];
  const _Float16* p = ph + b * N_ + lane * 32;
  float s = 0.f;
  for (int i = 0; i < 32; ++i)
    if ((bits >> i) & 1u) s += (float)p[i];
  s = wred_sum(s);
  if (lane == 0) denom[wid] = s;
}

// ---------------------------------------------------------------------------
// K5: edge aggregation GEMM:  edge[b] = att_edge @ x
// 4 waves/WG share one LDS-staged, double-buffered B panel (8KB/k-chunk);
// next panel is fetched to registers during compute, committed between barriers.
// A = maskbit * p[n] built branchlessly; per-row 1/denom + se = edge.v3 epilogue.
// ---------------------------------------------------------------------------
__global__ void __launch_bounds__(128)
k_edge(const _Float16* __restrict__ xT, const _Float16* __restrict__ ph,
       const unsigned* __restrict__ mask, const float* __restrict__ denom,
       const float* __restrict__ vbuf, _Float16* __restrict__ edgeT,
       float* __restrict__ se) {
  __shared__ uint4 lds4[2][F_ * 5];          // 2 x 128 rows x 80B (padded) = 20KB
  int tid  = threadIdx.x;
  int wave = tid >> 5, lane = tid & 31;
  int b    = blockIdx.x >> 3;                       // 8 quads per batch
  int e0   = ((blockIdx.x & 7) * 4 + wave) * 16;
  int c  = (lane < 16) ? 0 : 8;     // K-offset for this half of the wave
  int lr = lane & 15;               // A-matrix row held by this lane
  bool uf = (denom[b * E_ + e0 + lr] == 0.f);   // empty edge -> uniform softmax
  const unsigned*  mrp = mask + ((size_t)b * E_ + e0 + lr) * 32;
  const _Float16*  pb  = ph + b * N_;
  const _Float16*  xb  = xT + (size_t)b * F_ * N_;

  v8f zero = {};
  v8f acc[8];
#pragma unroll
  for (int t = 0; t < 8; ++t) acc[t] = zero;

  // preload panel 0 (row f = tid, 32 halves)
  {
    const uint4* src = (const uint4*)(xb + (size_t)tid * N_);
    uint4 r0 = src[0], r1 = src[1], r2 = src[2], r3 = src[3];
    lds4[0][tid * 5 + 0] = r0; lds4[0][tid * 5 + 1] = r1;
    lds4[0][tid * 5 + 2] = r2; lds4[0][tid * 5 + 3] = r3;
  }
  __syncthreads();

  for (int kc = 0; kc < N_ / 32; ++kc) {
    int n0 = kc * 32;
    // fetch next panel into registers while computing this one
    uint4 r0, r1, r2, r3;
    if (kc < N_ / 32 - 1) {
      const uint4* src = (const uint4*)(xb + (size_t)tid * N_ + n0 + 32);
      r0 = src[0]; r1 = src[1]; r2 = src[2]; r3 = src[3];
    }
    // ---- build A (p masked, branchless uniform fallback) ----
    unsigned bits = mrp[kc];
    uint4 alo = *(const uint4*)(pb + n0 + c);
    uint4 ahi = *(const uint4*)(pb + n0 + 16 + c);
    __builtin_prefetch(pb + n0 + 32 + c, 0, 0);
    unsigned aw[8] = {alo.x, alo.y, alo.z, alo.w, ahi.x, ahi.y, ahi.z, ahi.w};
    V16 A;
#pragma unroll
    for (int j = 0; j < 4; ++j) {
      unsigned p0 = aw[j]     & mask2(bits, c + 2 * j);
      unsigned p1 = aw[4 + j] & mask2(bits, 16 + c + 2 * j);
      ((unsigned*)&A)[j]     = uf ? 0x3C003C00u : p0;
      ((unsigned*)&A)[4 + j] = uf ? 0x3C003C00u : p1;
    }
    // ---- 8 WMMAs from the LDS panel ----
    const _Float16* bufp = (const _Float16*)lds4[kc & 1];
#pragma unroll
    for (int t = 0; t < 8; ++t) {
      const _Float16* col = bufp + (t * 16 + lr) * LDSROW + c;
      V16 Bv;
      Bv.u[0] = *(const uint4*)(col);
      Bv.u[1] = *(const uint4*)(col + 16);
      acc[t] = __builtin_amdgcn_wmma_f32_16x16x32_f16(false, A.h, false, Bv.h,
                                                      (short)0, acc[t], false, false);
    }
    // ---- commit next panel ----
    if (kc < N_ / 32 - 1) {
      __syncthreads();
      int nb = (kc + 1) & 1;
      lds4[nb][tid * 5 + 0] = r0; lds4[nb][tid * 5 + 1] = r1;
      lds4[nb][tid * 5 + 2] = r2; lds4[nb][tid * 5 + 3] = r3;
      __syncthreads();
    }
  }

  // epilogue: scale rows by 1/denom, emit edgeT (f16) and se = edge . v3
  float v3c[8];
#pragma unroll
  for (int t = 0; t < 8; ++t) v3c[t] = vbuf[2 * F_ + t * 16 + lr];
#pragma unroll
  for (int r = 0; r < 8; ++r) {
    int m = r + ((lane < 16) ? 0 : 8);
    int erow = e0 + m;
    float dm  = denom[b * E_ + erow];
    float scm = (dm == 0.f) ? (1.f / (float)N_) : (1.f / dm);
    float part = 0.f;
#pragma unroll
    for (int t = 0; t < 8; ++t) {
      float ev = acc[t][r] * scm;
      edgeT[(size_t)b * F_ * E_ + (size_t)(t * 16 + lr) * E_ + erow] = (_Float16)ev;
      part += ev * v3c[t];
    }
    part += __shfl_xor(part, 1, 32); part += __shfl_xor(part, 2, 32);
    part += __shfl_xor(part, 4, 32); part += __shfl_xor(part, 8, 32);
    if (lr == 0) se[b * E_ + erow] = part;
  }
}

// ---------------------------------------------------------------------------
// K6: node-level softmax stats per (b,n): max & 1/Z over masked lrelu(sn+se)
// nscale < 0 flags "no edges" -> uniform 1/E fallback
// ---------------------------------------------------------------------------
__global__ void k_nstat(const unsigned* __restrict__ maskT, const float* __restrict__ sn,
                        const float* __restrict__ se, float* __restrict__ nmax,
                        float* __restrict__ nscale) {
  int wid  = (blockIdx.x * blockDim.x + threadIdx.x) >> 5;
  int lane = threadIdx.x & 31;
  if (wid >= B_ * N_) return;
  int b = wid / N_;
  float snv = sn[wid];
  const float*    seb = se + b * E_;
  const unsigned* mw  = maskT + (size_t)wid * (E_ / 32);
  unsigned bits = (lane < 16) ? mw[lane] : 0u;
  float mx = -1e30f;
  if (lane < 16) {
    for (int i = 0; i < 32; ++i)
      if ((bits >> i) & 1u) mx = fmaxf(mx, lrelu(snv + seb[lane * 32 + i]));
  }
  for (int o = 1; o < 16; o <<= 1) mx = fmaxf(mx, __shfl_xor(mx, o, 32));
  bool uf = (mx < -1e29f);
  float Z = 0.f;
  if (lane < 16 && !uf) {
    for (int i = 0; i < 32; ++i)
      if ((bits >> i) & 1u) Z += __expf(lrelu(snv + seb[lane * 32 + i]) - mx);
  }
  for (int o = 1; o < 16; o <<= 1) Z += __shfl_xor(Z, o, 32);
  if (lane == 0) {
    nmax[wid]   = uf ? 0.f : mx;
    nscale[wid] = uf ? -1.f : 1.f / Z;
  }
}

// ---------------------------------------------------------------------------
// K7: node aggregation GEMM + ELU:  out[b] = elu(att_node @ edge)
// Same LDS-staged double-buffered B panel scheme as K5 (16 k-chunks of edgeT).
// ---------------------------------------------------------------------------
__global__ void __launch_bounds__(128)
k_node(const _Float16* __restrict__ edgeT, const unsigned* __restrict__ maskT,
       const float* __restrict__ sn, const float* __restrict__ se,
       const float* __restrict__ nmax, const float* __restrict__ nscale,
       float* __restrict__ out) {
  __shared__ uint4 lds4[2][F_ * 5];          // 2 x 128 rows x 80B (padded) = 20KB
  int tid  = threadIdx.x;
  int wave = tid >> 5, lane = tid & 31;
  int b    = blockIdx.x >> 4;                       // 16 quads per batch
  int n0   = ((blockIdx.x & 15) * 4 + wave) * 16;
  int c  = (lane < 16) ? 0 : 8;
  int lr = lane & 15;
  int nrow = n0 + lr;
  float snv  = sn[b * N_ + nrow];
  float mrow = nmax[b * N_ + nrow];
  bool  uf   = (nscale[b * N_ + nrow] < 0.f);
  const unsigned* mrp = maskT + ((size_t)b * N_ + nrow) * (E_ / 32);
  const float*    seb = se + b * E_;
  const _Float16* eb  = edgeT + (size_t)b * F_ * E_;

  v8f zero = {};
  v8f acc[8];
#pragma unroll
  for (int t = 0; t < 8; ++t) acc[t] = zero;

  // preload panel 0 (row f = tid, 32 halves of edgeT)
  {
    const uint4* src = (const uint4*)(eb + (size_t)tid * E_);
    uint4 r0 = src[0], r1 = src[1], r2 = src[2], r3 = src[3];
    lds4[0][tid * 5 + 0] = r0; lds4[0][tid * 5 + 1] = r1;
    lds4[0][tid * 5 + 2] = r2; lds4[0][tid * 5 + 3] = r3;
  }
  __syncthreads();

  for (int kc = 0; kc < E_ / 32; ++kc) {
    int e0k = kc * 32;
    uint4 r0, r1, r2, r3;
    if (kc < E_ / 32 - 1) {
      const uint4* src = (const uint4*)(eb + (size_t)tid * E_ + e0k + 32);
      r0 = src[0]; r1 = src[1]; r2 = src[2]; r3 = src[3];
    }
    // ---- build A: exp(lrelu(sn+se)-max) masked, branchless uniform fallback ----
    unsigned bits = mrp[kc];
    v16h A;
#pragma unroll
    for (int j = 0; j < 16; ++j) {
      int K = (j < 8) ? (c + j) : (16 + c + (j - 8));
      float sev = seb[e0k + K];
      float w = uf ? 1.f
                   : (((bits >> K) & 1u) ? __expf(lrelu(snv + sev) - mrow) : 0.f);
      A[j] = (_Float16)w;
    }
    // ---- 8 WMMAs from the LDS panel ----
    const _Float16* bufp = (const _Float16*)lds4[kc & 1];
#pragma unroll
    for (int t = 0; t < 8; ++t) {
      const _Float16* col = bufp + (t * 16 + lr) * LDSROW + c;
      V16 Bv;
      Bv.u[0] = *(const uint4*)(col);
      Bv.u[1] = *(const uint4*)(col + 16);
      acc[t] = __builtin_amdgcn_wmma_f32_16x16x32_f16(false, A, false, Bv.h,
                                                      (short)0, acc[t], false, false);
    }
    // ---- commit next panel ----
    if (kc < E_ / 32 - 1) {
      __syncthreads();
      int nb = (kc + 1) & 1;
      lds4[nb][tid * 5 + 0] = r0; lds4[nb][tid * 5 + 1] = r1;
      lds4[nb][tid * 5 + 2] = r2; lds4[nb][tid * 5 + 3] = r3;
      __syncthreads();
    }
  }

  // epilogue: per-row 1/Z (or uniform 1/E), ELU, coalesced f32 stores
#pragma unroll
  for (int r = 0; r < 8; ++r) {
    int m = r + ((lane < 16) ? 0 : 8);
    float s   = nscale[b * N_ + n0 + m];
    float scm = (s < 0.f) ? (1.f / (float)E_) : s;
#pragma unroll
    for (int t = 0; t < 8; ++t) {
      float v = acc[t][r] * scm;
      v = (v > 0.f) ? v : (__expf(v) - 1.f);   // ELU
      out[((size_t)b * N_ + n0 + m) * F_ + t * 16 + lr] = v;
    }
  }
}

// ---------------------------------------------------------------------------
extern "C" void kernel_launch(void* const* d_in, const int* in_sizes, int n_in,
                              void* d_out, int out_size, void* d_ws, size_t ws_size,
                              hipStream_t stream) {
  (void)in_sizes; (void)n_in; (void)out_size; (void)ws_size;
  const int*   inputs = (const int*)d_in[0];
  const int*   HT     = (const int*)d_in[1];
  const float* emb    = (const float*)d_in[2];
  const float* W2     = (const float*)d_in[3];
  const float* W3     = (const float*)d_in[4];
  const float* wc     = (const float*)d_in[5];
  const float* a1     = (const float*)d_in[6];
  const float* a2     = (const float*)d_in[7];
  float* out = (float*)d_out;

  char* p = (char*)d_ws;
  auto alloc = [&](size_t bytes) -> void* {
    void* r = (void*)p;
    p += (bytes + 255) & ~(size_t)255;
    return r;
  };
  float*     vbuf   = (float*)alloc(512 * sizeof(float));
  float*     s1     = (float*)alloc((size_t)B_ * N_ * sizeof(float));
  float*     sn     = (float*)alloc((size_t)B_ * N_ * sizeof(float));
  _Float16*  ph     = (_Float16*)alloc((size_t)B_ * N_ * sizeof(_Float16));
  float*     denom  = (float*)alloc((size_t)B_ * E_ * sizeof(float));
  float*     se     = (float*)alloc((size_t)B_ * E_ * sizeof(float));
  float*     nmax   = (float*)alloc((size_t)B_ * N_ * sizeof(float));
  float*     nscale = (float*)alloc((size_t)B_ * N_ * sizeof(float));
  unsigned*  mask   = (unsigned*)alloc((size_t)B_ * E_ * 32 * sizeof(unsigned));
  unsigned*  maskT  = (unsigned*)alloc((size_t)B_ * N_ * (E_ / 32) * sizeof(unsigned));
  _Float16*  xT     = (_Float16*)alloc((size_t)B_ * F_ * N_ * sizeof(_Float16));
  _Float16*  edgeT  = (_Float16*)alloc((size_t)B_ * F_ * E_ * sizeof(_Float16));

  k_prep<<<1, 128, 0, stream>>>(W2, W3, wc, a1, a2, vbuf);
  k_gather<<<(B_ * N_) / 8, 256, 0, stream>>>(inputs, emb, vbuf, xT, s1, sn);
  k_pmax<<<B_, 256, 0, stream>>>(s1, ph);
  k_mask<<<dim3(B_, E_ / 32), 1024, 0, stream>>>(HT, mask, maskT);
  k_denom<<<(B_ * E_) / 8, 256, 0, stream>>>(mask, ph, denom);
  k_edge<<<(B_ * (E_ / 16)) / 4, 128, 0, stream>>>(xT, ph, mask, denom, vbuf, edgeT, se);
  k_nstat<<<(B_ * N_) / 8, 256, 0, stream>>>(maskT, sn, se, nmax, nscale);
  k_node<<<(B_ * (N_ / 16)) / 4, 128, 0, stream>>>(edgeT, maskT, sn, se, nmax, nscale, out);
}
